// NeuroSAT_2705829397336
// MI455X (gfx1250) — matrix operand order repaired and checked by reading, not compiled, not run on hardware
//
#include <hip/hip_runtime.h>
#include <hip/hip_bf16.h>

// ---------------------------------------------------------------------------
// NeuroSAT message passing for MI455X (gfx1250, wave32).
// All dense matmuls run on v_wmma_f32_16x16x32_bf16 (bf16 inputs, f32 acc).
// Sparse scatter uses hardware global_atomic_add_f32.
// ---------------------------------------------------------------------------

#define DIMD 128
static constexpr int NV    = 50000;    // n_vars
static constexpr int NL    = 100000;   // n_lits
static constexpr int NC    = 210000;   // n_clauses
static constexpr int NCELL = 630000;   // n_cells
static constexpr int NP    = 100;      // n_probs
static constexpr int NR    = 8;        // n_rounds

typedef __attribute__((ext_vector_type(16))) __bf16 v16bf;
typedef __attribute__((ext_vector_type(8)))  __bf16 v8bf;
typedef __attribute__((ext_vector_type(8)))  float  v8f;

// ---------------------------------------------------------------------------
// WMMA GEMM:  out[N,M] = act( A[N,K](bf16) @ W[M,K]^T(bf16) + bias[M] )
// Block: 256 threads = 8 waves. Each wave: one 16x16 output tile.
// Block tile: 16 rows x 128 cols.  grid = (N/16, M/128).
// A fragment (16x32 bf16): lane<16 holds row (lane&15), K = kb+{0..7,16..23};
//                          lane>=16 same row, K = kb+{8..15,24..31}.
// B fragment (32x16 bf16): lane<16 holds col (lane&15), K = kb+{0..7,16..23};
//                          lane>=16 same col, K = kb+{8..15,24..31}.
// C/D (16x16 f32): VGPR r -> row r + (lane>=16 ? 8 : 0), col = lane&15.
// ---------------------------------------------------------------------------
template<bool RELU, bool OUT_BF16>
__global__ __launch_bounds__(256)
void gemm_wmma(const __bf16* __restrict__ A, const __bf16* __restrict__ W,
               const float* __restrict__ bias,
               float* __restrict__ outF, __bf16* __restrict__ outB,
               int N, int K, int M)
{
    const int lane    = threadIdx.x & 31;
    const int wave    = threadIdx.x >> 5;
    const int rowbase = blockIdx.x << 4;
    const int colbase = (blockIdx.y << 7) + (wave << 4);
    const int hi      = (lane >= 16) ? 1 : 0;
    const int arow    = rowbase + (lane & 15);
    const int col     = colbase + (lane & 15);

    const __bf16* Ap = A + (size_t)arow * K;
    const __bf16* Wp = W + (size_t)col  * K;

    v8f acc = {0.f, 0.f, 0.f, 0.f, 0.f, 0.f, 0.f, 0.f};

    for (int kb = 0; kb < K; kb += 32) {
        const int ka = kb + (hi ? 8 : 0);
        v8bf alo = *(const v8bf*)(Ap + ka);
        v8bf ahi = *(const v8bf*)(Ap + ka + 16);
        v8bf blo = *(const v8bf*)(Wp + ka);
        v8bf bhi = *(const v8bf*)(Wp + ka + 16);
        v16bf bv;
#pragma unroll
        for (int i = 0; i < 8; ++i) { bv[i] = blo[i]; bv[i + 8] = bhi[i]; }
        v16bf av;
#pragma unroll
        for (int i = 0; i < 8; ++i) { av[i] = alo[i]; av[i + 8] = ahi[i]; }
        acc = __builtin_amdgcn_wmma_f32_16x16x32_bf16(
                  /*neg_a=*/false, av, /*neg_b=*/false, bv,
                  /*c_mod=*/(short)0, acc, /*reuse_a=*/false, /*reuse_b=*/false);
    }

    const float bcol = bias[col];
#pragma unroll
    for (int r = 0; r < 8; ++r) {
        const int row = rowbase + r + (hi ? 8 : 0);
        float v = acc[r] + bcol;
        if (RELU) v = fmaxf(v, 0.f);
        const size_t o = (size_t)row * M + col;
        if (OUT_BF16) outB[o] = (__bf16)v;
        else          outF[o] = v;
    }
}

// -------------------------- elementwise helpers ----------------------------
__global__ void f32_to_bf16_k(const float* __restrict__ in, __bf16* __restrict__ out, int n)
{
    int i = blockIdx.x * 256 + threadIdx.x;
    if (i < n) out[i] = (__bf16)in[i];
}

__global__ void init_state_k(const float* __restrict__ W, const float* __restrict__ B,
                             float* __restrict__ h, float* __restrict__ c, int n)
{
    int i = blockIdx.x * 256 + threadIdx.x;
    if (i < n) { int d = i & (DIMD - 1); h[i] = W[d] + B[d]; c[i] = 0.f; }
}

// pack [x | h] -> bf16 [n,256]
__global__ void pack2_k(const float* __restrict__ x0, const float* __restrict__ x1,
                        __bf16* __restrict__ out, int n)
{
    int i = blockIdx.x * 256 + threadIdx.x;
    if (i >= n * 256) return;
    int row = i >> 8, col = i & 255;
    float v = (col < 128) ? x0[(size_t)row * 128 + col]
                          : x1[(size_t)row * 128 + (col - 128)];
    out[i] = (__bf16)v;
}

// pack [CL | L_h(flipped) | L_h] -> bf16 [nL,384]
__global__ void pack3L_k(const float* __restrict__ CL, const float* __restrict__ Lh,
                         __bf16* __restrict__ out, int nL, int nVars)
{
    int i = blockIdx.x * 256 + threadIdx.x;
    if (i >= nL * 384) return;
    int row = i / 384, col = i - row * 384;
    float v;
    if (col < 128)       v = CL[(size_t)row * 128 + col];
    else if (col < 256) {
        int fr = (row >= nVars) ? (row - nVars) : (row + nVars);
        v = Lh[(size_t)fr * 128 + (col - 128)];
    } else               v = Lh[(size_t)row * 128 + (col - 256)];
    out[i] = (__bf16)v;
}

// dst[didx[e],:] += src[sidx[e],:]   (128 lanes per cell, coalesced FP atomics)
__global__ void scatter_add_k(const float* __restrict__ src, const int* __restrict__ sidx,
                              const int* __restrict__ didx, float* __restrict__ dst, int ncell)
{
    int cell = blockIdx.x * 2 + (threadIdx.x >> 7);
    if (cell >= ncell) return;
    int d = threadIdx.x & 127;
    float v = src[(size_t)sidx[cell] * 128 + d];
    __hip_atomic_fetch_add(&dst[(size_t)didx[cell] * 128 + d], v,
                           __ATOMIC_RELAXED, __HIP_MEMORY_SCOPE_AGENT);
}

// ---------------- fused LSTM cell + LayerNorm(h) + LayerNorm(c) ------------
__device__ inline float sigm(float x) { return 1.f / (1.f + __expf(-x)); }
__device__ inline float wred(float v)
{
#pragma unroll
    for (int o = 16; o; o >>= 1) v += __shfl_xor(v, o, 32);
    return v;
}

__global__ __launch_bounds__(256)
void lstm_ln_k(const float* __restrict__ gates, float* __restrict__ h, float* __restrict__ c,
               const float* __restrict__ g1, const float* __restrict__ b1,
               const float* __restrict__ g2, const float* __restrict__ b2, int n)
{
    int row = blockIdx.x * 8 + (threadIdx.x >> 5);
    if (row >= n) return;
    int lane = threadIdx.x & 31;
    const float4* gp = (const float4*)(gates + (size_t)row * 512);
    float4 i4 = gp[lane];           // gate i : d = 4*lane..4*lane+3
    float4 f4 = gp[32 + lane];      // gate f
    float4 t4 = gp[64 + lane];      // gate g
    float4 o4 = gp[96 + lane];      // gate o
    float4 c4 = ((const float4*)(c + (size_t)row * 128))[lane];

    float iv[4] = {i4.x, i4.y, i4.z, i4.w};
    float fv[4] = {f4.x, f4.y, f4.z, f4.w};
    float tv[4] = {t4.x, t4.y, t4.z, t4.w};
    float ov[4] = {o4.x, o4.y, o4.z, o4.w};
    float cv[4] = {c4.x, c4.y, c4.z, c4.w};

    float hn[4], cn[4], sh = 0.f, shs = 0.f, sc = 0.f, scs = 0.f;
#pragma unroll
    for (int j = 0; j < 4; ++j) {
        float cc = sigm(fv[j]) * cv[j] + sigm(iv[j]) * tanhf(tv[j]);
        float hh = sigm(ov[j]) * tanhf(cc);
        cn[j] = cc; hn[j] = hh;
        sh += hh; shs += hh * hh; sc += cc; scs += cc * cc;
    }
    sh = wred(sh); shs = wred(shs); sc = wred(sc); scs = wred(scs);
    const float inv = 1.f / 128.f;
    float mh = sh * inv, vh = shs * inv - mh * mh;
    float mc = sc * inv, vc = scs * inv - mc * mc;
    float rh = rsqrtf(vh + 1e-5f), rc = rsqrtf(vc + 1e-5f);

    float4 ho, co;
    float hw[4], cw[4];
#pragma unroll
    for (int j = 0; j < 4; ++j) {
        int d = lane * 4 + j;
        hw[j] = (hn[j] - mh) * rh * g1[d] + b1[d];
        cw[j] = (cn[j] - mc) * rc * g2[d] + b2[d];
    }
    ho = make_float4(hw[0], hw[1], hw[2], hw[3]);
    co = make_float4(cw[0], cw[1], cw[2], cw[3]);
    ((float4*)(h + (size_t)row * 128))[lane] = ho;
    ((float4*)(c + (size_t)row * 128))[lane] = co;
}

// ------------------------ weight prep (per-launch) -------------------------
__global__ void build_wcatC_k(const float* __restrict__ Wih, const float* __restrict__ Whh,
                              __bf16* __restrict__ out)   // [512,256]
{
    int i = blockIdx.x * 256 + threadIdx.x;
    if (i >= 512 * 256) return;
    int o = i >> 8, k = i & 255;
    float v = (k < 128) ? Wih[o * 128 + k] : Whh[o * 128 + (k - 128)];
    out[i] = (__bf16)v;
}
__global__ void build_wcatL_k(const float* __restrict__ Wih, const float* __restrict__ Whh,
                              __bf16* __restrict__ out)   // [512,384]
{
    int i = blockIdx.x * 256 + threadIdx.x;
    if (i >= 512 * 384) return;
    int o = i / 384, k = i - o * 384;
    float v = (k < 256) ? Wih[o * 256 + k] : Whh[o * 128 + (k - 256)];
    out[i] = (__bf16)v;
}
__global__ void add_bias_k(const float* a, const float* b, float* out, int n)
{
    int i = blockIdx.x * 256 + threadIdx.x;
    if (i < n) out[i] = a[i] + b[i];
}

// ------------------------- final per-problem mean --------------------------
__global__ void prob_accum_k(const float* __restrict__ Lh, const int* __restrict__ pidx,
                             float* __restrict__ out, float* __restrict__ cnt, int nL)
{
    int lit = blockIdx.x * 2 + (threadIdx.x >> 7);
    if (lit >= nL) return;
    int d = threadIdx.x & 127;
    int p = pidx[lit];
    __hip_atomic_fetch_add(&out[(size_t)p * 128 + d], Lh[(size_t)lit * 128 + d],
                           __ATOMIC_RELAXED, __HIP_MEMORY_SCOPE_AGENT);
    if (d == 0)
        __hip_atomic_fetch_add(&cnt[p], 1.f, __ATOMIC_RELAXED, __HIP_MEMORY_SCOPE_AGENT);
}
__global__ void prob_div_k(float* out, const float* cnt, int n)
{
    int i = blockIdx.x * 256 + threadIdx.x;
    if (i < n) out[i] /= cnt[i >> 7];
}

// ---------------------------------------------------------------------------
extern "C" void kernel_launch(void* const* d_in, const int* in_sizes, int n_in,
                              void* d_out, int out_size, void* d_ws, size_t ws_size,
                              hipStream_t stream)
{
    // params flatten in JAX tree order (dict keys sorted alphabetically):
    const float* C_Whh    = (const float*)d_in[0];
    const float* C_Wih    = (const float*)d_in[1];
    const float* C_bhh    = (const float*)d_in[2];
    const float* C_bih    = (const float*)d_in[3];
    const float* C_init_W = (const float*)d_in[4];
    const float* C_init_b = (const float*)d_in[5];
    const float* C_ln1_b  = (const float*)d_in[6];
    const float* C_ln1_g  = (const float*)d_in[7];
    const float* C_ln2_b  = (const float*)d_in[8];
    const float* C_ln2_g  = (const float*)d_in[9];
    const float* C_msg_W1 = (const float*)d_in[10];
    const float* C_msg_W2 = (const float*)d_in[11];
    const float* C_msg_W3 = (const float*)d_in[12];
    const float* C_msg_b1 = (const float*)d_in[13];
    const float* C_msg_b2 = (const float*)d_in[14];
    const float* C_msg_b3 = (const float*)d_in[15];
    const float* L_Whh    = (const float*)d_in[16];
    const float* L_Wih    = (const float*)d_in[17];
    const float* L_bhh    = (const float*)d_in[18];
    const float* L_bih    = (const float*)d_in[19];
    const float* L_init_W = (const float*)d_in[20];
    const float* L_init_b = (const float*)d_in[21];
    const float* L_ln1_b  = (const float*)d_in[22];
    const float* L_ln1_g  = (const float*)d_in[23];
    const float* L_ln2_b  = (const float*)d_in[24];
    const float* L_ln2_g  = (const float*)d_in[25];
    const float* L_msg_W1 = (const float*)d_in[26];
    const float* L_msg_W2 = (const float*)d_in[27];
    const float* L_msg_W3 = (const float*)d_in[28];
    const float* L_msg_b1 = (const float*)d_in[29];
    const float* L_msg_b2 = (const float*)d_in[30];
    const float* L_msg_b3 = (const float*)d_in[31];
    const int*   lit_idx    = (const int*)d_in[32];
    const int*   clause_idx = (const int*)d_in[33];
    const int*   prob_idx   = (const int*)d_in[34];

    // ---- workspace carve (256-byte aligned) ----
    char* wsp = (char*)d_ws;
    auto carve = [&wsp](size_t bytes) -> void* {
        void* p = wsp; wsp += (bytes + 255) & ~(size_t)255; return p;
    };
    float*  L_h    = (float*) carve((size_t)NL * 128 * 4);
    float*  L_c    = (float*) carve((size_t)NL * 128 * 4);
    float*  C_h    = (float*) carve((size_t)NC * 128 * 4);
    float*  C_c    = (float*) carve((size_t)NC * 128 * 4);
    float*  LC     = (float*) carve((size_t)NC * 128 * 4);
    float*  CL     = (float*) carve((size_t)NL * 128 * 4);
    float*  pre    = (float*) carve((size_t)NC * 128 * 4);        // max(NL,NC)x128
    float*  gates  = (float*) carve((size_t)NC * 512 * 4);        // max rows
    __bf16* bufA   = (__bf16*)carve((size_t)NC * 128 * 2);
    __bf16* bufB   = (__bf16*)carve((size_t)NC * 128 * 2);
    __bf16* packed = (__bf16*)carve((size_t)NC * 256 * 2);        // >= NL*384
    __bf16* wLm[3], *wCm[3];
    for (int i = 0; i < 3; ++i) wLm[i] = (__bf16*)carve(16384 * 2);
    for (int i = 0; i < 3; ++i) wCm[i] = (__bf16*)carve(16384 * 2);
    __bf16* wcatC  = (__bf16*)carve(512 * 256 * 2);
    __bf16* wcatL  = (__bf16*)carve(512 * 384 * 2);
    float*  biasC  = (float*) carve(512 * 4);
    float*  biasL  = (float*) carve(512 * 4);
    float*  counts = (float*) carve(NP * 4);
    float*  out    = (float*)d_out;                               // [NP,128]

    const dim3 T256(256);

    // ---- per-launch weight prep (deterministic) ----
    const float* LmsgW[3] = {L_msg_W1, L_msg_W2, L_msg_W3};
    const float* CmsgW[3] = {C_msg_W1, C_msg_W2, C_msg_W3};
    for (int i = 0; i < 3; ++i) {
        f32_to_bf16_k<<<(16384 + 255) / 256, T256, 0, stream>>>(LmsgW[i], wLm[i], 16384);
        f32_to_bf16_k<<<(16384 + 255) / 256, T256, 0, stream>>>(CmsgW[i], wCm[i], 16384);
    }
    build_wcatC_k<<<(512 * 256 + 255) / 256, T256, 0, stream>>>(C_Wih, C_Whh, wcatC);
    build_wcatL_k<<<(512 * 384 + 255) / 256, T256, 0, stream>>>(L_Wih, L_Whh, wcatL);
    add_bias_k<<<2, T256, 0, stream>>>(C_bih, C_bhh, biasC, 512);
    add_bias_k<<<2, T256, 0, stream>>>(L_bih, L_bhh, biasL, 512);

    // ---- initial state ----
    init_state_k<<<(NL * 128 + 255) / 256, T256, 0, stream>>>(L_init_W, L_init_b, L_h, L_c, NL * 128);
    init_state_k<<<(NC * 128 + 255) / 256, T256, 0, stream>>>(C_init_W, C_init_b, C_h, C_c, NC * 128);

    const float* LmsgB[3] = {L_msg_b1, L_msg_b2, L_msg_b3};
    const float* CmsgB[3] = {C_msg_b1, C_msg_b2, C_msg_b3};

    const dim3 gL16(NL / 16, 1), gC16(NC / 16, 1);     // GEMM M=128
    const dim3 gLg(NL / 16, 4),  gCg(NC / 16, 4);      // GEMM M=512 (gates)

    for (int r = 0; r < NR; ++r) {
        // ---- literal -> clause ----
        f32_to_bf16_k<<<(NL * 128 + 255) / 256, T256, 0, stream>>>(L_h, bufA, NL * 128);
        gemm_wmma<true,  true ><<<gL16, T256, 0, stream>>>(bufA, wLm[0], LmsgB[0], nullptr, bufB, NL, 128, 128);
        gemm_wmma<true,  true ><<<gL16, T256, 0, stream>>>(bufB, wLm[1], LmsgB[1], nullptr, bufA, NL, 128, 128);
        gemm_wmma<false, false><<<gL16, T256, 0, stream>>>(bufA, wLm[2], LmsgB[2], pre, nullptr, NL, 128, 128);

        hipMemsetAsync(LC, 0, (size_t)NC * 128 * 4, stream);
        scatter_add_k<<<(NCELL + 1) / 2, T256, 0, stream>>>(pre, lit_idx, clause_idx, LC, NCELL);

        pack2_k<<<(NC * 256 + 255) / 256, T256, 0, stream>>>(LC, C_h, packed, NC);
        gemm_wmma<false, false><<<gCg, T256, 0, stream>>>(packed, wcatC, biasC, gates, nullptr, NC, 256, 512);
        lstm_ln_k<<<(NC + 7) / 8, T256, 0, stream>>>(gates, C_h, C_c,
                                                     C_ln1_g, C_ln1_b, C_ln2_g, C_ln2_b, NC);

        // ---- clause -> literal ----
        f32_to_bf16_k<<<(NC * 128 + 255) / 256, T256, 0, stream>>>(C_h, bufA, NC * 128);
        gemm_wmma<true,  true ><<<gC16, T256, 0, stream>>>(bufA, wCm[0], CmsgB[0], nullptr, bufB, NC, 128, 128);
        gemm_wmma<true,  true ><<<gC16, T256, 0, stream>>>(bufB, wCm[1], CmsgB[1], nullptr, bufA, NC, 128, 128);
        gemm_wmma<false, false><<<gC16, T256, 0, stream>>>(bufA, wCm[2], CmsgB[2], pre, nullptr, NC, 128, 128);

        hipMemsetAsync(CL, 0, (size_t)NL * 128 * 4, stream);
        scatter_add_k<<<(NCELL + 1) / 2, T256, 0, stream>>>(pre, clause_idx, lit_idx, CL, NCELL);

        pack3L_k<<<(NL * 384 + 255) / 256, T256, 0, stream>>>(CL, L_h, packed, NL, NV);
        gemm_wmma<false, false><<<gLg, T256, 0, stream>>>(packed, wcatL, biasL, gates, nullptr, NL, 384, 512);
        lstm_ln_k<<<(NL + 7) / 8, T256, 0, stream>>>(gates, L_h, L_c,
                                                     L_ln1_g, L_ln1_b, L_ln2_g, L_ln2_b, NL);
    }

    // ---- per-problem mean of literal embeddings ----
    hipMemsetAsync(out, 0, (size_t)NP * 128 * 4, stream);
    hipMemsetAsync(counts, 0, (size_t)NP * 4, stream);
    prob_accum_k<<<(NL + 1) / 2, T256, 0, stream>>>(L_h, prob_idx, out, counts, NL);
    prob_div_k<<<(NP * 128 + 255) / 256, T256, 0, stream>>>(out, counts, NP * 128);
}
